// QuantizedEncoder_72782515798907
// MI455X (gfx1250) — matrix operand order repaired and checked by reading, not compiled, not run on hardware
//
#include <hip/hip_runtime.h>
#include <hip/hip_bf16.h>

#define CIN   256
#define Tlen  600000
#define Llen  (Tlen / 5)
#define ROWS  32                   // frames per fused tile iteration
#define TILES (Llen / ROWS)        // 3750 tiles of 32 frames
#define FTILE (Llen / 16)          // first-conv tiles (16 frames)
#define HSTR  264                  // padded LDS stride (halfs), multiple of 8 -> 16B aligned frags
#define WSTR  264

typedef _Float16 h16;
typedef __attribute__((ext_vector_type(16))) _Float16 v16h;
typedef __attribute__((ext_vector_type(8)))  _Float16 v8h;
typedef __attribute__((ext_vector_type(8)))  float    v8f;

union V16U { v16h v; v8h h[2]; };

// LDS layout (in halfs)
#define OFF_WPW 0
#define OFF_WLC (256 * WSTR)
#define OFF_H   (2 * 256 * WSTR)
#define OFF_P   (2 * 256 * WSTR + ROWS * HSTR)
#define SMEM_HALFS (2 * 256 * WSTR + 2 * ROWS * HSTR)
#define SMEM_BYTES (SMEM_HALFS * 2)          // 304128 B  (< 320KB WGP LDS)

__device__ __forceinline__ float leaky(float x) { return x >= 0.f ? x : 0.3f * x; }
__device__ __forceinline__ float clampq(float x) { return fminf(fmaxf(x, -128.f), 127.f); }

// A fragment: 16-bit A 16x32 layout. lane: row=lane&15, kh=lane>>4.
// elems 0..7  -> k = k0 + 8*kh + j        (contiguous 16B)
// elems 8..15 -> k = k0 + 16 + 8*kh + j   (contiguous 16B)
__device__ __forceinline__ v16h load_frag_a(const h16* sA, int row, int kh, int k0) {
    const h16* p = sA + row * HSTR + k0 + 8 * kh;
    V16U u;
    u.h[0] = *(const v8h*)(p);
    u.h[1] = *(const v8h*)(p + 16);
    return u.v;
}

// B fragment: 32x16 B layout from n-major (transposed) weights.
// lane: n = ntile + (lane&15), kh = lane>>4; elems 0..7 -> k = k0 + 8*kh + j, elems 8..15 -> k = k0 + 16 + 8*kh + j
__device__ __forceinline__ v16h load_frag_b(const h16* sW, int n, int kh, int k0) {
    const h16* p = sW + n * WSTR + k0 + 8 * kh;
    V16U u;
    u.h[0] = *(const v8h*)(p);
    u.h[1] = *(const v8h*)(p + 16);
    return u.v;
}

#define WMMA_F16(a, b, c) \
    __builtin_amdgcn_wmma_f32_16x16x32_f16(false, (a), false, (b), (short)0, (c), false, false)

// 2(M)x2(N) register-tiled GEMM over a 32x256 LDS activation tile.
// 4 WMMAs per k-step share 2 A + 2 B fragment loads (1 ds-frag per WMMA);
// next-step fragments are loaded before current-step WMMAs issue.
__device__ __forceinline__ void gemm_tile32(const h16* __restrict__ sA,
                                            const h16* __restrict__ sW,
                                            int row, int kh, int n0,
                                            v8f& acc00, v8f& acc01,
                                            v8f& acc10, v8f& acc11) {
    const h16* sA1 = sA + 16 * HSTR;     // second M-tile (rows 16..31)
    v16h a0 = load_frag_a(sA,  row, kh, 0);
    v16h a1 = load_frag_a(sA1, row, kh, 0);
    v16h b0 = load_frag_b(sW, n0 + row,      kh, 0);
    v16h b1 = load_frag_b(sW, n0 + 16 + row, kh, 0);
#pragma unroll
    for (int kk = 0; kk < 7; ++kk) {
        const int k0n = (kk + 1) * 32;
        v16h a0n = load_frag_a(sA,  row, kh, k0n);
        v16h a1n = load_frag_a(sA1, row, kh, k0n);
        v16h b0n = load_frag_b(sW, n0 + row,      kh, k0n);
        v16h b1n = load_frag_b(sW, n0 + 16 + row, kh, k0n);
        acc00 = WMMA_F16(a0, b0, acc00);
        acc01 = WMMA_F16(a0, b1, acc01);
        acc10 = WMMA_F16(a1, b0, acc10);
        acc11 = WMMA_F16(a1, b1, acc11);
        a0 = a0n; a1 = a1n; b0 = b0n; b1 = b1n;
    }
    acc00 = WMMA_F16(a0, b0, acc00);
    acc01 = WMMA_F16(a0, b1, acc01);
    acc10 = WMMA_F16(a1, b0, acc10);
    acc11 = WMMA_F16(a1, b1, acc11);
}

// ---------------------------------------------------------------------------
// Stage 1: first conv  (k=10, stride=5, 1 -> 256 ch) + bias.  res1[t][c]
// ---------------------------------------------------------------------------
__global__ __launch_bounds__(256) void k_first(const float* __restrict__ x,
                                               const float* __restrict__ fbuf,   // [5]
                                               const float* __restrict__ wf,     // [10][256]
                                               const float* __restrict__ bf,     // [256]
                                               float* __restrict__ res1) {
    const int c  = threadIdx.x;
    const int t0 = blockIdx.x * 16;
    float w[10];
#pragma unroll
    for (int k = 0; k < 10; ++k) w[k] = wf[k * CIN + c];
    const float b = bf[c];
#pragma unroll
    for (int r = 0; r < 16; ++r) {
        const int t = t0 + r;
        const int base = 5 * t;                 // index into padded x (len T+5)
        float acc = b;
#pragma unroll
        for (int k = 0; k < 10; ++k) {
            const int i = base + k;
            const float xv = (i < 5) ? fbuf[i] : x[i - 5];   // uniform -> s_load
            acc = fmaf(xv, w[k], acc);
        }
        res1[(size_t)t * CIN + c] = acc;
    }
}

// ---------------------------------------------------------------------------
// Fused residual block: dw(k3,dil) -> pw GEMM -> [quant] -> leaky -> lc GEMM
//                       -> epilogue (dequant / residual / requant)
// BLK = 1,2,3 selects the per-block quantization math.
// ---------------------------------------------------------------------------
template <int BLK, int DIL>
__global__ __launch_bounds__(256, 1) void k_resblock(const float* __restrict__ resPrev,
                                                     const float* __restrict__ padbuf,  // [2*DIL][256]
                                                     const float* __restrict__ w_dw,    // [3][256]
                                                     const float* __restrict__ b_dw,
                                                     const float* __restrict__ w_pw,    // [256][256] (k-major)
                                                     const float* __restrict__ b_pw,
                                                     const float* __restrict__ w_lc,
                                                     const float* __restrict__ b_lc,
                                                     float* __restrict__ resNext) {
    extern __shared__ h16 smem[];
    h16* sWpw = smem + OFF_WPW;
    h16* sWlc = smem + OFF_WLC;
    h16* sH   = smem + OFF_H;
    h16* sP   = smem + OFF_P;

    const int tid  = threadIdx.x;
    const int wid  = tid >> 5;
    const int lane = tid & 31;
    const int row  = lane & 15;      // A row-in-tile / B,C,D column-in-tile
    const int kh   = lane >> 4;

    // Load + transpose weights to f16 n-major in LDS (coalesced global reads).
    for (int k = 0; k < CIN; ++k) {
        sWpw[tid * WSTR + k] = (h16)w_pw[k * CIN + tid];
        sWlc[tid * WSTR + k] = (h16)w_lc[k * CIN + tid];
    }
    const float wd0 = w_dw[tid], wd1 = w_dw[CIN + tid], wd2 = w_dw[2 * CIN + tid];
    const float bd  = b_dw[tid];
    const int n0 = wid * 32;
    const float bpw0 = b_pw[n0 + row],      bpw1 = b_pw[n0 + 16 + row];
    const float blc0 = b_lc[n0 + row],      blc1 = b_lc[n0 + 16 + row];

    constexpr int NR = ROWS + 2 * DIL;

    for (int tile = blockIdx.x; tile < TILES; tile += gridDim.x) {
        const int t0 = tile * ROWS;
        __syncthreads();   // prev iteration fully done with sH/sP

        // prefetch next persistent tile's activations (global_prefetch_b8)
        {
            const int nt = tile + gridDim.x;
            if (nt < TILES) {
                int g = nt * ROWS - 2 * DIL; if (g < 0) g = 0;
                __builtin_prefetch(resPrev + (size_t)g * CIN + tid, 0, 0);
            }
        }

        // ---- depthwise conv (thread = channel) -> sH (f16 A tile 32x256) ----
        float vbuf[NR];
#pragma unroll
        for (int i = 0; i < NR; ++i) {
            const int pi = t0 + i;            // padded-domain row
            const int g  = pi - 2 * DIL;      // global frame
            float v;
            if (g < 0) {
                const float bv = padbuf[pi * CIN + tid];
                if constexpr (BLK == 1)      v = bv;                                        // raw ci buffer
                else if constexpr (BLK == 2) v = clampq(rintf(bv / 5.548006534576416f - 47.f));
                else                         v = clampq(rintf(bv / 4.458680152893066f - 38.f));
            } else {
                const float rv = resPrev[(size_t)g * CIN + tid];
                if constexpr (BLK == 1)      v = leaky(rv);                                 // ci = leaky(res)
                else                         v = clampq(rintf(leaky(rv)));                  // q(dq(leaky)) == clip(round(leaky))
            }
            vbuf[i] = v;
        }
#pragma unroll
        for (int r = 0; r < ROWS; ++r) {
            const float hh = fmaf(vbuf[r], wd0, fmaf(vbuf[r + DIL], wd1,
                             fmaf(vbuf[r + 2 * DIL], wd2, bd)));
            sH[r * HSTR + tid] = (h16)hh;
        }
        __syncthreads();

        // ---- GEMM1: sH[32x256] x Wpw -> quant/leaky -> sP (f16) ----
        {
            v8f a00 = {}, a01 = {}, a10 = {}, a11 = {};
            gemm_tile32(sH, sWpw, row, kh, n0, a00, a01, a10, a11);
#pragma unroll
            for (int v = 0; v < 8; ++v) {
                const int m = v + 8 * kh;
                float y00 = a00[v] + bpw0, y01 = a01[v] + bpw1;
                float y10 = a10[v] + bpw0, y11 = a11[v] + bpw1;
                if constexpr (BLK == 1) {   // qp=(17.6297, -67) then leaky
                    y00 = leaky(clampq(rintf(y00 / 17.62967872619629f + 67.f)));
                    y01 = leaky(clampq(rintf(y01 / 17.62967872619629f + 67.f)));
                    y10 = leaky(clampq(rintf(y10 / 17.62967872619629f + 67.f)));
                    y11 = leaky(clampq(rintf(y11 / 17.62967872619629f + 67.f)));
                } else {
                    y00 = leaky(y00); y01 = leaky(y01);
                    y10 = leaky(y10); y11 = leaky(y11);
                }
                sP[m * HSTR + n0 + row]             = (h16)y00;
                sP[m * HSTR + n0 + 16 + row]        = (h16)y01;
                sP[(m + 16) * HSTR + n0 + row]      = (h16)y10;
                sP[(m + 16) * HSTR + n0 + 16 + row] = (h16)y11;
            }
        }
        __syncthreads();

        // ---- GEMM2: sP x Wlc -> epilogue -> global ----
        {
            v8f a00 = {}, a01 = {}, a10 = {}, a11 = {};
            gemm_tile32(sP, sWlc, row, kh, n0, a00, a01, a10, a11);
            const int c0 = n0 + row, c1 = n0 + 16 + row;
#pragma unroll
            for (int v = 0; v < 8; ++v) {
                const int m  = v + 8 * kh;
                const int tA = t0 + m;          // M-tile 0
                const int tB = t0 + 16 + m;     // M-tile 1
                const float rpA0 = resPrev[(size_t)tA * CIN + c0];
                const float rpA1 = resPrev[(size_t)tA * CIN + c1];
                const float rpB0 = resPrev[(size_t)tB * CIN + c0];
                const float rpB1 = resPrev[(size_t)tB * CIN + c1];
                float oA0, oA1, oB0, oB1;
                if constexpr (BLK == 1) {
                    // ro = dequant(lc+b, 6.52806, -16); res2 = quant(res1+ro, 12.7165, -23)
                    const float rA0 = (a00[v] + blc0 - 16.f) * 6.528060436248779f;
                    const float rA1 = (a01[v] + blc1 - 16.f) * 6.528060436248779f;
                    const float rB0 = (a10[v] + blc0 - 16.f) * 6.528060436248779f;
                    const float rB1 = (a11[v] + blc1 - 16.f) * 6.528060436248779f;
                    oA0 = clampq(rintf((rpA0 + rA0) / 12.716455459594727f + 23.f));
                    oA1 = clampq(rintf((rpA1 + rA1) / 12.716455459594727f + 23.f));
                    oB0 = clampq(rintf((rpB0 + rB0) / 12.716455459594727f + 23.f));
                    oB1 = clampq(rintf((rpB1 + rB1) / 12.716455459594727f + 23.f));
                } else if constexpr (BLK == 2) {
                    oA0 = a00[v] + blc0 + rpA0;
                    oA1 = a01[v] + blc1 + rpA1;
                    oB0 = a10[v] + blc0 + rpB0;
                    oB1 = a11[v] + blc1 + rpB1;
                } else {
                    oA0 = (leaky(a00[v] + blc0 + rpA0) + 34.f) * 3.698859930038452f;
                    oA1 = (leaky(a01[v] + blc1 + rpA1) + 34.f) * 3.698859930038452f;
                    oB0 = (leaky(a10[v] + blc0 + rpB0) + 34.f) * 3.698859930038452f;
                    oB1 = (leaky(a11[v] + blc1 + rpB1) + 34.f) * 3.698859930038452f;
                }
                resNext[(size_t)tA * CIN + c0] = oA0;
                resNext[(size_t)tA * CIN + c1] = oA1;
                resNext[(size_t)tB * CIN + c0] = oB0;
                resNext[(size_t)tB * CIN + c1] = oB1;
            }
        }
    }
}

// ---------------------------------------------------------------------------
extern "C" void kernel_launch(void* const* d_in, const int* in_sizes, int n_in,
                              void* d_out, int out_size, void* d_ws, size_t ws_size,
                              hipStream_t stream) {
    const float* x     = (const float*)d_in[0];
    const float* fbuf  = (const float*)d_in[1];
    const float* r1buf = (const float*)d_in[2];
    const float* r2buf = (const float*)d_in[3];
    const float* r3buf = (const float*)d_in[4];
    const float* wf    = (const float*)d_in[5];
    const float* bf    = (const float*)d_in[6];
    const float* wdw1 = (const float*)d_in[7],  *bdw1 = (const float*)d_in[8];
    const float* wpw1 = (const float*)d_in[9],  *bpw1 = (const float*)d_in[10];
    const float* wlc1 = (const float*)d_in[11], *blc1 = (const float*)d_in[12];
    const float* wdw2 = (const float*)d_in[13], *bdw2 = (const float*)d_in[14];
    const float* wpw2 = (const float*)d_in[15], *bpw2 = (const float*)d_in[16];
    const float* wlc2 = (const float*)d_in[17], *blc2 = (const float*)d_in[18];
    const float* wdw3 = (const float*)d_in[19], *bdw3 = (const float*)d_in[20];
    const float* wpw3 = (const float*)d_in[21], *bpw3 = (const float*)d_in[22];
    const float* wlc3 = (const float*)d_in[23], *blc3 = (const float*)d_in[24];
    float* out = (float*)d_out;

    float* resA = (float*)d_ws;                       // ping: res1, then res3
    float* resB = resA + (size_t)Llen * CIN;          // pong: res2 (quantized)

    k_first<<<FTILE, 256, 0, stream>>>(x, fbuf, wf, bf, resA);

    const int NWG = 512;    // persistent workgroups; weights loaded to LDS once each
    k_resblock<1, 1><<<NWG, 256, SMEM_BYTES, stream>>>(resA, r1buf, wdw1, bdw1, wpw1, bpw1, wlc1, blc1, resB);
    k_resblock<2, 3><<<NWG, 256, SMEM_BYTES, stream>>>(resB, r2buf, wdw2, bdw2, wpw2, bpw2, wlc2, blc2, resA);
    k_resblock<3, 9><<<NWG, 256, SMEM_BYTES, stream>>>(resA, r3buf, wdw3, bdw3, wpw3, bpw3, wlc3, blc3, out);
}
